// DefaultModel_17093969838684
// MI455X (gfx1250) — matrix operand order repaired and checked by reading, not compiled, not run on hardware
//
#include <hip/hip_runtime.h>
#include <hip/hip_bf16.h>

typedef __attribute__((ext_vector_type(16))) _Float16 v16h;
typedef __attribute__((ext_vector_type(8)))  _Float16 v8h;
typedef __attribute__((ext_vector_type(8)))  float    v8f;

#define BN_EPS 1e-5f

// ---------------------------------------------------------------------------
// conv0: 3->1, k3 pad1, +bias, no BN/ReLU. Input = concat(left,right) over batch.
__global__ void conv0_kernel(const float* __restrict__ left, const float* __restrict__ right,
                             const float* __restrict__ w, const float* __restrict__ bias,
                             float* __restrict__ out, int H, int W) {
  long idx = (long)blockIdx.x * blockDim.x + threadIdx.x;
  long total = 16L * H * W;
  if (idx >= total) return;
  int x = (int)(idx % W); long t = idx / W;
  int y = (int)(t % H); int b = (int)(t / H);
  const float* src = (b < 8) ? (left + (size_t)b * 3 * H * W)
                             : (right + (size_t)(b - 8) * 3 * H * W);
  float acc = bias[0];
  for (int ci = 0; ci < 3; ++ci)
    for (int r = 0; r < 3; ++r) {
      int ih = y + r - 1;
      if (ih < 0 || ih >= H) continue;
      for (int s = 0; s < 3; ++s) {
        int iw = x + s - 1;
        if (iw < 0 || iw >= W) continue;
        acc += src[((size_t)ci * H + ih) * W + iw] * w[(ci * 3 + r) * 3 + s];
      }
    }
  out[idx] = acc;
}

// ---------------------------------------------------------------------------
// small conv (k=3, pad=1) + BN + ReLU, tiny channel counts
__global__ void nconv_small_kernel(const float* __restrict__ in, float* __restrict__ out,
                                   const float* __restrict__ w, const float* __restrict__ g,
                                   const float* __restrict__ bb, const float* __restrict__ m,
                                   const float* __restrict__ v,
                                   int Bn, int Cin, int Cout, int Hin, int Win, int stride) {
  int Hout = (Hin - 1) / stride + 1;   // pad=1, k=3
  int Wout = (Win - 1) / stride + 1;
  long idx = (long)blockIdx.x * blockDim.x + threadIdx.x;
  long total = (long)Bn * Cout * Hout * Wout;
  if (idx >= total) return;
  int ow = (int)(idx % Wout); long t = idx / Wout;
  int oh = (int)(t % Hout); t /= Hout;
  int co = (int)(t % Cout); int b = (int)(t / Cout);
  float acc = 0.f;
  for (int ci = 0; ci < Cin; ++ci) {
    const float* ip = in + (size_t)(b * Cin + ci) * Hin * Win;
    const float* wp = w + (size_t)(co * Cin + ci) * 9;
    for (int r = 0; r < 3; ++r) {
      int ih = oh * stride + r - 1;
      if (ih < 0 || ih >= Hin) continue;
      for (int s = 0; s < 3; ++s) {
        int iw = ow * stride + s - 1;
        if (iw < 0 || iw >= Win) continue;
        acc += ip[(size_t)ih * Win + iw] * wp[r * 3 + s];
      }
    }
  }
  float sc = g[co] * rsqrtf(v[co] + BN_EPS);
  out[idx] = fmaxf((acc - m[co]) * sc + bb[co], 0.f);
}

// ---------------------------------------------------------------------------
__global__ void maxpool_kernel(const float* __restrict__ in, float* __restrict__ out,
                               int BC, int Hin, int Win) {
  int Hout = Hin >> 1, Wout = Win >> 1;
  long idx = (long)blockIdx.x * blockDim.x + threadIdx.x;
  long total = (long)BC * Hout * Wout;
  if (idx >= total) return;
  int ow = (int)(idx % Wout); long t = idx / Wout;
  int oh = (int)(t % Hout); int bc = (int)(t / Hout);
  const float* ip = in + (size_t)bc * Hin * Win;
  int y = oh * 2, x = ow * 2;
  float a = ip[(size_t)y * Win + x];
  float b = ip[(size_t)y * Win + x + 1];
  float c = ip[(size_t)(y + 1) * Win + x];
  float d = ip[(size_t)(y + 1) * Win + x + 1];
  out[idx] = fmaxf(fmaxf(a, b), fmaxf(c, d));
}

// ---------------------------------------------------------------------------
// bilinear resize matching bilinear_ac (align-corners style linspace)
__global__ void bilinear_kernel(const float* __restrict__ in, float* __restrict__ out,
                                int BC, int Hin, int Win, int Hout, int Wout) {
  long idx = (long)blockIdx.x * blockDim.x + threadIdx.x;
  long total = (long)BC * Hout * Wout;
  if (idx >= total) return;
  int ox = (int)(idx % Wout); long t = idx / Wout;
  int oy = (int)(t % Hout); int bc = (int)(t / Hout);
  float fy = (Hout > 1) ? (float)oy * (float)(Hin - 1) / (float)(Hout - 1) : 0.f;
  float fx = (Wout > 1) ? (float)ox * (float)(Win - 1) / (float)(Wout - 1) : 0.f;
  int y0 = (int)floorf(fy); y0 = y0 < 0 ? 0 : (y0 > Hin - 1 ? Hin - 1 : y0);
  int x0 = (int)floorf(fx); x0 = x0 < 0 ? 0 : (x0 > Win - 1 ? Win - 1 : x0);
  int y1 = y0 + 1 < Hin ? y0 + 1 : Hin - 1;
  int x1 = x0 + 1 < Win ? x0 + 1 : Win - 1;
  float wy = fy - (float)y0, wx = fx - (float)x0;
  const float* ip = in + (size_t)bc * Hin * Win;
  float v00 = ip[(size_t)y0 * Win + x0], v01 = ip[(size_t)y0 * Win + x1];
  float v10 = ip[(size_t)y1 * Win + x0], v11 = ip[(size_t)y1 * Win + x1];
  out[idx] = (v00 * (1.f - wx) + v01 * wx) * (1.f - wy)
           + (v10 * (1.f - wx) + v11 * wx) * wy;
}

// ---------------------------------------------------------------------------
__global__ void concat_kernel(const float* __restrict__ a, const float* __restrict__ b,
                              float* __restrict__ out, int Bn, int Ca, int Cb, int HW) {
  long idx = (long)blockIdx.x * blockDim.x + threadIdx.x;
  int Ct = Ca + Cb;
  long total = (long)Bn * Ct * HW;
  if (idx >= total) return;
  int p = (int)(idx % HW); long t = idx / HW;
  int c = (int)(t % Ct); int bn = (int)(t / Ct);
  out[idx] = (c < Ca) ? a[((size_t)bn * Ca + c) * HW + p]
                      : b[((size_t)bn * Cb + (c - Ca)) * HW + p];
}

// ---------------------------------------------------------------------------
// cost volume: feat (16,8,H,W) -> cost (8,24,H,W)
__global__ void cost_kernel(const float* __restrict__ feat, float* __restrict__ out,
                            int H, int W) {
  long idx = (long)blockIdx.x * blockDim.x + threadIdx.x;
  long total = 8L * 24 * H * W;
  if (idx >= total) return;
  int x = (int)(idx % W); long t = idx / W;
  int y = (int)(t % H); t /= H;
  int d = (int)(t % 24); int b = (int)(t / 24);
  int ix = x - d;
  float s = 0.f;
  for (int c = 0; c < 8; ++c) {
    float l = feat[(((size_t)b * 8 + c) * H + y) * W + x];
    float r = (ix >= 0) ? feat[(((size_t)(b + 8) * 8 + c) * H + y) * W + ix] : 0.f;
    s += fabsf(l - r);
  }
  out[idx] = s;
}

// ---------------------------------------------------------------------------
// softmax(-c2) expectation over 24 disparities, scaled by 8 (=544/68)
__global__ void softargmin_kernel(const float* __restrict__ c2, float* __restrict__ out,
                                  int H, int W) {
  long idx = (long)blockIdx.x * blockDim.x + threadIdx.x;
  long total = 8L * H * W;
  if (idx >= total) return;
  int x = (int)(idx % W); long t = idx / W;
  int y = (int)(t % H); int b = (int)(t / H);
  const float* p = c2 + (size_t)b * 24 * H * W + (size_t)y * W + x;
  size_t cs = (size_t)H * W;
  float mx = -3.0e38f;
  for (int d = 0; d < 24; ++d) mx = fmaxf(mx, -p[(size_t)d * cs]);
  float se = 0.f, sw = 0.f;
  for (int d = 0; d < 24; ++d) {
    float e = expf(-p[(size_t)d * cs] - mx);
    se += e; sw += e * (float)d;
  }
  out[idx] = (sw / se) * 8.0f;
}

// ---------------------------------------------------------------------------
// Implicit-GEMM conv (stride 1, pad K/2) + BN + ReLU via V_WMMA_F32_16X16X32_F16.
// Fully compile-time shapes: CIN/COUT template params fold all guards and the
// N-tile count, so the K-loop is pure ds_load_b128 pairs + back-to-back WMMAs.
// Tile = 16 consecutive pixels of one image row; im2col tile staged in LDS as
// f16 [16 px][KPAD] (coalesced global loads); weights staged once per block.
//   A frag: lane<16 row=lane  K={kA0..kA0+7, kA0+16..kA0+23}, lane>=16 kA0+=8
//   B frag: lane<16 col=lane  K={kB0..kB0+7, kB0+16..kB0+23}, lane>=16 kB0+=8
template <int K, int KSTEPS, int WPB, int CIN, int COUT>
__global__ void __launch_bounds__(WPB * 32)
wmma_conv_kernel(const float* __restrict__ in, float* __restrict__ out,
                 const float* __restrict__ w, const float* __restrict__ g,
                 const float* __restrict__ bb, const float* __restrict__ m,
                 const float* __restrict__ v,
                 int Bn, int H, int W) {
  constexpr int KK = K * K;
  constexpr int pad = K >> 1;
  constexpr int Ktot = CIN * KK;
  constexpr int KPAD = KSTEPS * 32;
  constexpr int RS = KPAD + 8;   // +8 halves: 16 staging lanes hit 16 distinct banks
  constexpr int NTL = (COUT + 15) >> 4;
  static_assert(KSTEPS == (Ktot + 31) / 32, "KSTEPS mismatch");
  __shared__ __align__(16) _Float16 ldsA[WPB * 16 * RS];
  __shared__ __align__(16) _Float16 ldsB[32 * RS];

  const int lane = threadIdx.x & 31;
  const int wslot = threadIdx.x >> 5;

  // ---- stage weights (B matrix, zero-padded) into LDS, shared by the block
  for (int idx = threadIdx.x; idx < 32 * KPAD; idx += WPB * 32) {
    int n = idx / KPAD, k = idx - n * KPAD;
    float val = (n < COUT && k < Ktot) ? w[(size_t)n * Ktot + k] : 0.f;
    ldsB[n * RS + k] = (_Float16)val;
  }
  __syncthreads();

  const int tpr = (W + 15) >> 4;                 // pixel tiles per image row
  const long ntiles = (long)Bn * H * tpr;
  const long wave = (long)blockIdx.x * WPB + wslot;
  if (wave >= ntiles) return;                    // wave-uniform: EXEC all-ones below

  const int tx = (int)(wave % tpr);
  const long rowid = wave / tpr;
  const int oh = (int)(rowid % H);
  const int b = (int)(rowid / H);
  const int ow0 = tx * 16;

  // ---- stage im2col A tile: lanes = (k parity) x (16 pixels), coalesced in px
  {
    const int px = lane & 15;
    const int kg = lane >> 4;
    _Float16* Arow = ldsA + (wslot * 16 + px) * RS;
    const float* ibase = in + (size_t)b * CIN * H * W;
    for (int kk = 0; kk < KPAD; kk += 2) {
      int k = kk + kg;
      float val = 0.f;
      if (k < Ktot) {
        int ci = k / KK; int rs = k - ci * KK;
        int r = rs / K;  int s = rs - r * K;
        int ih = oh + r - pad;
        int iw = ow0 + px + s - pad;
        if (ih >= 0 && ih < H && iw >= 0 && iw < W)
          val = ibase[((size_t)ci * H + ih) * W + iw];
      }
      Arow[k] = (_Float16)val;   // same-wave DS ops stay in order
    }
  }

  const int hi = lane >> 4;
  const int ncol = lane & 15;
  const _Float16* Ap = ldsA + (size_t)(wslot * 16 + (lane & 15)) * RS;
  v8f acc0 = {0.f, 0.f, 0.f, 0.f, 0.f, 0.f, 0.f, 0.f};
  v8f acc1 = acc0;

  for (int t = 0; t < KSTEPS; ++t) {
    union { v16h vv; v8h h[2]; } A;
    const int kA0 = t * 32 + (hi ? 8 : 0);
    A.h[0] = *(const v8h*)(Ap + kA0);        // 16B aligned
    A.h[1] = *(const v8h*)(Ap + kA0 + 16);
    const int kB0 = t * 32 + (hi ? 8 : 0);
    {
      const _Float16* Bp = ldsB + (size_t)ncol * RS + kB0;
      union { v16h vv; v8h h[2]; } Bm;
      Bm.h[0] = *(const v8h*)(Bp);
      Bm.h[1] = *(const v8h*)(Bp + 16);
      acc0 = __builtin_amdgcn_wmma_f32_16x16x32_f16(false, A.vv, false, Bm.vv,
                                                    (short)0, acc0, false, false);
    }
    if (NTL > 1) {                            // compile-time constant -> folds
      const _Float16* Bp = ldsB + (size_t)(ncol + 16) * RS + kB0;
      union { v16h vv; v8h h[2]; } Bm;
      Bm.h[0] = *(const v8h*)(Bp);
      Bm.h[1] = *(const v8h*)(Bp + 16);
      acc1 = __builtin_amdgcn_wmma_f32_16x16x32_f16(false, A.vv, false, Bm.vv,
                                                    (short)0, acc1, false, false);
    }
  }

  // ---- epilogue: D element r -> (pixel ow0 + r + 8*hi, channel ncol [+16])
  const int mbase = hi << 3;
#pragma unroll
  for (int nt = 0; nt < NTL; ++nt) {
    int n = ncol + nt * 16;
    if (n < COUT) {
      float sc = g[n] * rsqrtf(v[n] + BN_EPS);
      float bi = bb[n] - m[n] * sc;
      v8f acc = nt ? acc1 : acc0;
      float* obase = out + ((size_t)(b * COUT + n) * H + oh) * W;
#pragma unroll
      for (int r = 0; r < 8; ++r) {
        int ow = ow0 + mbase + r;
        if (ow < W)
          obase[ow] = fmaxf(acc[r] * sc + bi, 0.f);
      }
    }
  }
}

// ===========================================================================
extern "C" void kernel_launch(void* const* d_in, const int* in_sizes, int n_in,
                              void* d_out, int out_size, void* d_ws, size_t ws_size,
                              hipStream_t stream) {
  (void)n_in; (void)out_size; (void)ws_size;
  auto F = [&](int i) { return (const float*)d_in[i]; };
  struct Layer { const float *w, *g, *b, *m, *v; };
  const float *left, *right, *conv0_w, *conv0_b;
  Layer nconv1, d1[2], d2[2], d3[2], fupL[2], pdownL, pn[2], pupL[2], plastL;
  {
    int i = 0;
    if (in_sizes[0] == in_sizes[1]) {
      // insertion-order flattening: left,right, conv0_w,conv0_b, layers as (w,g,b,m,v)
      left = F(i++); right = F(i++); conv0_w = F(i++); conv0_b = F(i++);
      auto rd = [&](Layer& L) { L.w = F(i++); L.g = F(i++); L.b = F(i++); L.m = F(i++); L.v = F(i++); };
      rd(nconv1); rd(d1[0]); rd(d1[1]); rd(d2[0]); rd(d2[1]); rd(d3[0]); rd(d3[1]);
      rd(fupL[0]); rd(fupL[1]); rd(pdownL); rd(pn[0]); rd(pn[1]); rd(pupL[0]); rd(pupL[1]); rd(plastL);
    } else {
      // jax pytree sorted-key flattening: left, {sorted params}, right; layers as (b,g,m,v,w)
      left = F(i++); conv0_b = F(i++); conv0_w = F(i++);
      auto rd = [&](Layer& L) { L.b = F(i++); L.g = F(i++); L.m = F(i++); L.v = F(i++); L.w = F(i++); };
      rd(d1[0]); rd(d1[1]); rd(d2[0]); rd(d2[1]); rd(d3[0]); rd(d3[1]);
      rd(fupL[0]); rd(fupL[1]); rd(nconv1); rd(pdownL); rd(plastL);
      rd(pn[0]); rd(pn[1]); rd(pupL[0]); rd(pupL[1]);
      right = F(i++);
    }
  }

  const int H0 = 544, W0 = 960;
  const int H1 = 272, W1 = 480;
  const int H2 = 136, W2 = 240;
  const int H3 = 68,  W3 = 120;
  const int H4 = 34,  W4 = 60;

  // workspace layout (floats), with manual reuse; peak = 10,967,040 floats (~42 MB)
  float* ws = (float*)d_ws;
  const size_t A_conv0 = 0;                 // 8,355,840
  const size_t A_x1    = 8355840;           // 2,088,960
  const size_t A_p1    = 0;
  const size_t A_d1a   = 522240;
  const size_t A_x2    = 1566720;
  const size_t A_p2    = 2611200;
  const size_t A_d2a   = 2872320;
  const size_t A_x3    = 3394560;           // kept for concat
  const size_t A_p3    = 3916800;
  const size_t A_d3a   = 4047360;
  const size_t A_x4    = 4308480;
  const size_t A_u     = 4569600;
  const size_t A_cat1  = 5614080;
  const size_t A_f1    = 7180800;           // ends 8,225,280
  const size_t A_feat  = 8355840;           // x1 dead
  const size_t A_cost  = 9400320;           // kept for concat2
  const size_t A_pc    = 0;                 // front region free again
  const size_t A_pd    = 391680;
  const size_t A_pn1   = 783360;
  const size_t A_c1    = 1175040;
  const size_t A_cu    = 1566720;
  const size_t A_cat2  = 3133440;
  const size_t A_pup1  = 6266880;           // ends 7,833,600
  const size_t A_pup2  = 0;
  const size_t A_c2    = 1566720;           // cu dead
  const size_t A_pred  = 3133440;           // cat2 dead

  auto blk = [](long n) { return (unsigned)((n + 255) / 256); };
  auto wtiles = [](int Bn, int H, int W) { return (long)Bn * H * ((W + 15) / 16); };

  auto sconv = [&](const float* in, float* out, const Layer& L, int Bn, int Cin, int Cout,
                   int Hin, int Win, int stride) {
    int Ho = (Hin - 1) / stride + 1, Wo = (Win - 1) / stride + 1;
    nconv_small_kernel<<<blk((long)Bn * Cout * Ho * Wo), 256, 0, stream>>>(
        in, out, L.w, L.g, L.b, L.m, L.v, Bn, Cin, Cout, Hin, Win, stride);
  };
  // WMMA conv launchers; KSTEPS = ceil(Cin*K*K/32), WPB keeps LDS < 64KB
  auto wc48_24 = [&](const float* in, float* out, const Layer& L, int Bn, int H, int W) {
    long nt = wtiles(Bn, H, W);
    wmma_conv_kernel<3, 14, 2, 48, 24><<<(unsigned)((nt + 1) / 2), 64, 0, stream>>>(
        in, out, L.w, L.g, L.b, L.m, L.v, Bn, H, W);
  };
  auto wc24_24 = [&](const float* in, float* out, const Layer& L, int Bn, int H, int W) {
    long nt = wtiles(Bn, H, W);
    wmma_conv_kernel<3, 7, 4, 24, 24><<<(unsigned)((nt + 3) / 4), 128, 0, stream>>>(
        in, out, L.w, L.g, L.b, L.m, L.v, Bn, H, W);
  };
  auto wc24_24_1x1 = [&](const float* in, float* out, const Layer& L, int Bn, int H, int W) {
    long nt = wtiles(Bn, H, W);
    wmma_conv_kernel<1, 1, 8, 24, 24><<<(unsigned)((nt + 7) / 8), 256, 0, stream>>>(
        in, out, L.w, L.g, L.b, L.m, L.v, Bn, H, W);
  };
  auto wc12_8 = [&](const float* in, float* out, const Layer& L, int Bn, int H, int W) {
    long nt = wtiles(Bn, H, W);
    wmma_conv_kernel<3, 4, 8, 12, 8><<<(unsigned)((nt + 7) / 8), 256, 0, stream>>>(
        in, out, L.w, L.g, L.b, L.m, L.v, Bn, H, W);
  };
  auto wc8_8 = [&](const float* in, float* out, const Layer& L, int Bn, int H, int W) {
    long nt = wtiles(Bn, H, W);
    wmma_conv_kernel<3, 3, 8, 8, 8><<<(unsigned)((nt + 7) / 8), 256, 0, stream>>>(
        in, out, L.w, L.g, L.b, L.m, L.v, Bn, H, W);
  };

  // 1) conv0 (3->1) on concat batch
  conv0_kernel<<<blk(16L * H0 * W0), 256, 0, stream>>>(left, right, conv0_w, conv0_b,
                                                       ws + A_conv0, H0, W0);
  // 2) nconv1 (1->1, stride 2)
  sconv(ws + A_conv0, ws + A_x1, nconv1, 16, 1, 1, H0, W0, 2);
  // 3) pool -> down1 (1->2, 2->2)
  maxpool_kernel<<<blk(16L * H2 * W2), 256, 0, stream>>>(ws + A_x1, ws + A_p1, 16, H1, W1);
  sconv(ws + A_p1,  ws + A_d1a, d1[0], 16, 1, 2, H2, W2, 1);
  sconv(ws + A_d1a, ws + A_x2,  d1[1], 16, 2, 2, H2, W2, 1);
  // 4) pool -> down2 (2->4, 4->4)
  maxpool_kernel<<<blk(16L * 2 * H3 * W3), 256, 0, stream>>>(ws + A_x2, ws + A_p2, 16 * 2, H2, W2);
  sconv(ws + A_p2,  ws + A_d2a, d2[0], 16, 2, 4, H3, W3, 1);
  sconv(ws + A_d2a, ws + A_x3,  d2[1], 16, 4, 4, H3, W3, 1);
  // 5) pool -> down3 (4->8, then 8->8 via WMMA)
  maxpool_kernel<<<blk(16L * 4 * H4 * W4), 256, 0, stream>>>(ws + A_x3, ws + A_p3, 16 * 4, H3, W3);
  sconv(ws + A_p3, ws + A_d3a, d3[0], 16, 4, 8, H4, W4, 1);
  wc8_8(ws + A_d3a, ws + A_x4, d3[1], 16, H4, W4);
  // 6) upsample x4, concat with x3, fup (12->8, 8->8) via WMMA
  bilinear_kernel<<<blk(16L * 8 * H3 * W3), 256, 0, stream>>>(ws + A_x4, ws + A_u,
                                                              16 * 8, H4, W4, H3, W3);
  concat_kernel<<<blk(16L * 12 * H3 * W3), 256, 0, stream>>>(ws + A_x3, ws + A_u, ws + A_cat1,
                                                             16, 4, 8, H3 * W3);
  wc12_8(ws + A_cat1, ws + A_f1,   fupL[0], 16, H3, W3);
  wc8_8 (ws + A_f1,   ws + A_feat, fupL[1], 16, H3, W3);
  // 7) cost volume (8,24,68,120)
  cost_kernel<<<blk(8L * 24 * H3 * W3), 256, 0, stream>>>(ws + A_feat, ws + A_cost, H3, W3);
  // 8) pool cost -> pdown (1x1, 24->24) -> pnconv x2 (24->24) via WMMA
  maxpool_kernel<<<blk(8L * 24 * H4 * W4), 256, 0, stream>>>(ws + A_cost, ws + A_pc, 8 * 24, H3, W3);
  wc24_24_1x1(ws + A_pc,  ws + A_pd,  pdownL, 8, H4, W4);
  wc24_24(ws + A_pd,  ws + A_pn1, pn[0],  8, H4, W4);
  wc24_24(ws + A_pn1, ws + A_c1,  pn[1],  8, H4, W4);
  // 9) upsample c1, concat with cost, pup (48->24, 24->24), plast (24->24) via WMMA
  bilinear_kernel<<<blk(8L * 24 * H3 * W3), 256, 0, stream>>>(ws + A_c1, ws + A_cu,
                                                              8 * 24, H4, W4, H3, W3);
  concat_kernel<<<blk(8L * 48 * H3 * W3), 256, 0, stream>>>(ws + A_cost, ws + A_cu, ws + A_cat2,
                                                            8, 24, 24, H3 * W3);
  wc48_24(ws + A_cat2, ws + A_pup1, pupL[0], 8, H3, W3);
  wc24_24(ws + A_pup1, ws + A_pup2, pupL[1], 8, H3, W3);
  wc24_24(ws + A_pup2, ws + A_c2,   plastL,  8, H3, W3);
  // 10) soft-argmin + x8 scale, then bilinear upsample to full res -> d_out
  softargmin_kernel<<<blk(8L * H3 * W3), 256, 0, stream>>>(ws + A_c2, ws + A_pred, H3, W3);
  bilinear_kernel<<<blk(8L * H0 * W0), 256, 0, stream>>>(ws + A_pred, (float*)d_out,
                                                         8, H3, W3, H0, W0);
}
